// Block_39479339384999
// MI455X (gfx1250) — matrix operand order repaired and checked by reading, not compiled, not run on hardware
//
#include <hip/hip_runtime.h>
#include <cmath>

// ---------------------------------------------------------------------------
// Types for gfx1250 WMMA (wave32, v_wmma_f32_16x16x32_bf16)
// ---------------------------------------------------------------------------
typedef __bf16 bf16;
typedef __attribute__((ext_vector_type(16))) __bf16       v16bf;
typedef __attribute__((ext_vector_type(8)))  float        v8f;
typedef __attribute__((ext_vector_type(4)))  unsigned int u32x4;

#define B_    32
#define N_    577
#define C_    768
#define H_    12
#define HD_   64
#define HID_  3072
#define NPAD  608          // 577 padded to 19*32
#define RT_   37           // ceil(577/16) row/col tiles
#define M1_   (B_ * N_)    // 18464 tokens (divisible by 16)
#define KEEP_ 404          // ceil(0.7*576)
#define N3_   405
#define M3_   (B_ * N3_)   // 12960 (divisible by 16)

union F16U { v16bf v; u32x4 u[2]; };

// A-fragment (16x32, MxK): lane holds row m=lane&15; halfs i: K = k0 + (i<8?i:i+8) + 8*(lane>=16)
__device__ __forceinline__ v16bf load_frag_a(const bf16* base, int ld, int k0, int lane) {
  const int m = lane & 15, s = lane >> 4;
  const bf16* p = base + (long)m * ld + k0 + 8 * s;
  F16U t;
  t.u[0] = *reinterpret_cast<const u32x4*>(p);
  t.u[1] = *reinterpret_cast<const u32x4*>(p + 16);
  return t.v;
}
// B-fragment (32x16, KxN): lane holds col n=lane&15; halfs i: K = k0 + (i<8?i:i+8) + 8*(lane>=16)
__device__ __forceinline__ v16bf load_frag_b(const bf16* base, int ld, int k0, int lane) {
  const int n = lane & 15, s = lane >> 4;
  const bf16* p = base + (long)n * ld + k0 + 8 * s;
  F16U t;
  t.u[0] = *reinterpret_cast<const u32x4*>(p);
  t.u[1] = *reinterpret_cast<const u32x4*>(p + 16);
  return t.v;
}
__device__ __forceinline__ v8f wmma_bf16(v16bf a, v16bf b, v8f c) {
  return __builtin_amdgcn_wmma_f32_16x16x32_bf16(false, a, false, b, (short)0, c, false, false);
}

// One wave computes a 16(M) x 64(N) fp32 tile: out = A(MxK) * W(NxK)^T
__device__ __forceinline__ void gemm_core_16x64(const bf16* A, int lda,
                                                const bf16* Bp, int ldb,
                                                int K, int lane, v8f acc[4]) {
  for (int k0 = 0; k0 < K; k0 += 32) {
    v16bf a = load_frag_a(A, lda, k0, lane);
    if (k0 + 128 < K) {   // emits global_prefetch_b8 (CDNA5)
      __builtin_prefetch(A  + (long)(lane & 15) * lda + k0 + 128, 0, 0);
      __builtin_prefetch(Bp + (long)(lane & 15) * ldb + k0 + 128, 0, 0);
    }
#pragma unroll
    for (int t = 0; t < 4; ++t) {
      v16bf b = load_frag_b(Bp + (long)(t * 16) * ldb, ldb, k0, lane);
      acc[t] = wmma_bf16(a, b, acc[t]);
    }
  }
}

__device__ __forceinline__ float block_sum256(float v, float* red) {
  const int t = threadIdx.x;
  red[t] = v; __syncthreads();
  for (int o = 128; o > 0; o >>= 1) { if (t < o) red[t] += red[t + o]; __syncthreads(); }
  float r = red[0]; __syncthreads();
  return r;
}

// ---------------------------------------------------------------------------
__global__ void cvt_bf16_kernel(const float* __restrict__ s, bf16* __restrict__ d, int n) {
  int i = blockIdx.x * 256 + threadIdx.x;
  const int stride = gridDim.x * 256;
  for (; i < n; i += stride) d[i] = (bf16)s[i];
}

__global__ void ln1_kernel(const float* __restrict__ x, const float* __restrict__ w,
                           const float* __restrict__ b, bf16* __restrict__ out) {
  __shared__ float red[256];
  const long row = blockIdx.x;
  const float* xr = x + row * C_;
  float s = 0.f;
  for (int c = threadIdx.x; c < C_; c += 256) s += xr[c];
  const float mu = block_sum256(s, red) * (1.f / C_);
  float v = 0.f;
  for (int c = threadIdx.x; c < C_; c += 256) { float d = xr[c] - mu; v += d * d; }
  const float inv = rsqrtf(block_sum256(v, red) * (1.f / C_) + 1e-5f);
  for (int c = threadIdx.x; c < C_; c += 256)
    out[row * C_ + c] = (bf16)((xr[c] - mu) * inv * w[c] + b[c]);
}

__global__ void mnmx_kernel(const float* __restrict__ pos, float* __restrict__ out_mm) {
  __shared__ float rmn[256], rmx[256];
  const int b = blockIdx.x, t = threadIdx.x;
  const float* p = pos + (long)b * N_ * 2 + 2;   // tokens 1..576
  float mn = 3.4e38f, mx = -3.4e38f;
  for (int q = t; q < 576 * 576; q += 256) {
    const int i = q / 576, j = q - i * 576;
    const float dx = p[i*2] - p[j*2], dy = p[i*2+1] - p[j*2+1];
    const float d = sqrtf(dx*dx + dy*dy + 1e-5f);
    mn = fminf(mn, d); mx = fmaxf(mx, d);
  }
  rmn[t] = mn; rmx[t] = mx; __syncthreads();
  for (int o = 128; o > 0; o >>= 1) {
    if (t < o) { rmn[t] = fminf(rmn[t], rmn[t+o]); rmx[t] = fmaxf(rmx[t], rmx[t+o]); }
    __syncthreads();
  }
  if (t == 0) { out_mm[b*2] = rmn[0]; out_mm[b*2+1] = rmx[0]; }
}

// QKV: xn(18464x768) @ qkv_w^T(768x2304) -> scatter bf16 q (pre-scaled), k, vT
__global__ void __launch_bounds__(128) gemm_qkv_kernel(const bf16* __restrict__ xn,
    const bf16* __restrict__ wb, const float* __restrict__ gamma,
    bf16* __restrict__ q, bf16* __restrict__ k, bf16* __restrict__ vT) {
  const int lane = threadIdx.x & 31;
  const int w = blockIdx.x * 4 + (threadIdx.x >> 5);
  const int MT = M1_ / 16;
  if (w >= MT * (3 * C_ / 64)) return;
  const int mt = w % MT, nt = w / MT;
  const v8f zero = {0,0,0,0,0,0,0,0};
  v8f acc[4] = {zero, zero, zero, zero};
  gemm_core_16x64(xn + (long)mt * 16 * C_, C_, wb + (long)nt * 64 * C_, C_, C_, lane, acc);
  const float g  = fminf(fmaxf(gamma[0], 0.f), 1.f);
  const float qs = (1.f - g) * 0.125f;          // (1-g) * hd^-0.5
  const int s = lane >> 4, n15 = lane & 15;
#pragma unroll
  for (int r = 0; r < 8; ++r) {
    const int m = mt * 16 + r + 8 * s;
    const int b = m / N_, nn = m - b * N_;
#pragma unroll
    for (int t = 0; t < 4; ++t) {
      const int n  = nt * 64 + t * 16 + n15;
      const int s3 = n / C_, rem = n - s3 * C_;
      const int hh = rem >> 6, d = rem & 63;
      const long bh = (long)b * H_ + hh;
      const float v = acc[t][r];
      if (s3 == 0)      q [(bh * NPAD + nn) * HD_ + d]   = (bf16)(v * qs);
      else if (s3 == 1) k [(bh * NPAD + nn) * HD_ + d]   = (bf16)v;
      else              vT[(bh * HD_  + d)  * NPAD + nn] = (bf16)v;
    }
  }
}

// Attention: one wave per (b, h, 16-row tile). Scores via WMMA -> LDS, exact softmax,
// P->bf16 in LDS, PV via WMMA. Also emits attn@pos per row and CLS probabilities.
__global__ void attn_kernel(const bf16* __restrict__ q, const bf16* __restrict__ k,
    const bf16* __restrict__ vT, const float* __restrict__ pos,
    const float* __restrict__ mm, const float* __restrict__ gamma,
    bf16* __restrict__ attn_out, float* __restrict__ pos_part,
    float* __restrict__ cls_part) {
  __shared__ float sc[16][NPAD];
  __shared__ bf16  pb[16][NPAD];
  __shared__ float srowm[16], srowinv[16];
  const int lane = threadIdx.x, s = lane >> 4, n15 = lane & 15;
  const int rt = blockIdx.x, h = blockIdx.y, b = blockIdx.z;
  const int r0 = rt * 16;
  const long bh = (long)b * H_ + h;
  const bf16* qb = q  + bh * NPAD * HD_;
  const bf16* kb = k  + bh * NPAD * HD_;
  const bf16* vb = vT + bh * HD_  * NPAD;
  const float* pp = pos + (long)b * N_ * 2;
  const float g = fminf(fmaxf(gamma[0], 0.f), 1.f);
  const float mn = mm[b*2], inv_rg = 1.f / (mm[b*2+1] - mn);

  const v16bf aq0 = load_frag_a(qb + (long)r0 * HD_, HD_, 0,  lane);
  const v16bf aq1 = load_frag_a(qb + (long)r0 * HD_, HD_, 32, lane);

  float prx[8], pry[8];
#pragma unroll
  for (int r = 0; r < 8; ++r) {
    int m = r0 + r + 8 * s; if (m > 576) m = 576;
    prx[r] = pp[m*2]; pry[r] = pp[m*2+1];
  }

  for (int ct = 0; ct < RT_; ++ct) {
    v8f sacc = {0,0,0,0,0,0,0,0};
    sacc = wmma_bf16(aq0, load_frag_b(kb + (long)ct*16*HD_, HD_, 0,  lane), sacc);
    sacc = wmma_bf16(aq1, load_frag_b(kb + (long)ct*16*HD_, HD_, 32, lane), sacc);
    const int j  = ct * 16 + n15;
    const int jc = j > 576 ? 576 : j;
    const float pjx = pp[jc*2], pjy = pp[jc*2+1];
#pragma unroll
    for (int r = 0; r < 8; ++r) {
      const int m = r0 + r + 8 * s;
      float logit;
      if (j >= N_) logit = -1e30f;
      else {
        float mk;
        if (j == 0 || m == 0) mk = 1.f;
        else {
          const float dx = prx[r] - pjx, dy = pry[r] - pjy;
          const float d = sqrtf(dx*dx + dy*dy + 1e-5f);
          mk = 1.f - (d - mn) * inv_rg;
        }
        logit = sacc[r] + g * mk;   // q pre-scaled by (1-g)*scale
      }
      sc[r + 8*s][j] = logit;
    }
  }
  __syncthreads();

  if (lane < 16) {
    const int row = lane;
    float mx = -3.4e38f;
    for (int jj = 0; jj < N_; ++jj) mx = fmaxf(mx, sc[row][jj]);
    float sum = 0.f, px = 0.f, py = 0.f;
    for (int jj = 0; jj < N_; ++jj) {
      const float e = expf(sc[row][jj] - mx);
      sum += e; px += e * pp[jj*2]; py += e * pp[jj*2+1];
    }
    const float inv = 1.f / sum;
    srowm[row] = mx; srowinv[row] = inv;
    const int gr = r0 + row;
    if (gr < N_) {
      pos_part[(bh * N_ + gr) * 2 + 0] = px * inv;
      pos_part[(bh * N_ + gr) * 2 + 1] = py * inv;
    }
    if (r0 == 0 && row == 0)
      for (int jj = 1; jj < N_; ++jj)
        cls_part[bh * (N_-1) + jj - 1] = expf(sc[0][jj] - mx) * inv;
  }
  __syncthreads();

  for (int row = 0; row < 16; ++row) {
    const float mx = srowm[row], inv = srowinv[row];
    for (int jj = lane; jj < NPAD; jj += 32)
      pb[row][jj] = (jj < N_) ? (bf16)(expf(sc[row][jj] - mx) * inv) : (bf16)0.f;
  }
  __syncthreads();

  const v8f zero = {0,0,0,0,0,0,0,0};
  v8f oacc[4] = {zero, zero, zero, zero};
  for (int k0 = 0; k0 < NPAD; k0 += 32) {
    const v16bf a = load_frag_a(&pb[0][0], NPAD, k0, lane);
#pragma unroll
    for (int t = 0; t < 4; ++t)
      oacc[t] = wmma_bf16(a, load_frag_b(vb + (long)(t*16)*NPAD, NPAD, k0, lane), oacc[t]);
  }
#pragma unroll
  for (int r = 0; r < 8; ++r) {
    const int m = r0 + r + 8 * s;
    if (m < N_) {
#pragma unroll
      for (int t = 0; t < 4; ++t)
        attn_out[((long)b * N_ + m) * C_ + h * HD_ + t * 16 + n15] = (bf16)oacc[t][r];
    }
  }
}

__global__ void __launch_bounds__(128) gemm_proj_kernel(const bf16* __restrict__ ao,
    const bf16* __restrict__ wb, const float* __restrict__ bias,
    const float* __restrict__ x, float* __restrict__ x2) {
  const int lane = threadIdx.x & 31;
  const int w = blockIdx.x * 4 + (threadIdx.x >> 5);
  const int MT = M1_ / 16;
  if (w >= MT * (C_ / 64)) return;
  const int mt = w % MT, nt = w / MT;
  const v8f zero = {0,0,0,0,0,0,0,0};
  v8f acc[4] = {zero, zero, zero, zero};
  gemm_core_16x64(ao + (long)mt * 16 * C_, C_, wb + (long)nt * 64 * C_, C_, C_, lane, acc);
  const int s = lane >> 4, n15 = lane & 15;
#pragma unroll
  for (int r = 0; r < 8; ++r) {
    const long m = mt * 16 + r + 8 * s;
#pragma unroll
    for (int t = 0; t < 4; ++t) {
      const int n = nt * 64 + t * 16 + n15;
      x2[m * C_ + n] = acc[t][r] + bias[n] + x[m * C_ + n];
    }
  }
}

__global__ void topk_kernel(const float* __restrict__ cls_part, int* __restrict__ idx) {
  __shared__ float vals[576];
  __shared__ float rv[256];
  __shared__ int   ri[256];
  const int b = blockIdx.x, t = threadIdx.x;
  for (int j = t; j < 576; j += 256) {
    float s = 0.f;
    for (int h = 0; h < H_; ++h) s += cls_part[((long)b*H_ + h)*576 + j];
    vals[j] = s * (1.f / H_);
  }
  __syncthreads();
  for (int it = 0; it < KEEP_; ++it) {
    float bv = -3.4e38f; int bi = 0x7fffffff;
    for (int j = t; j < 576; j += 256) {
      const float v = vals[j];
      if (v > bv) { bv = v; bi = j; }
    }
    rv[t] = bv; ri[t] = bi; __syncthreads();
    for (int o = 128; o > 0; o >>= 1) {
      if (t < o && (rv[t+o] > rv[t] || (rv[t+o] == rv[t] && ri[t+o] < ri[t]))) {
        rv[t] = rv[t+o]; ri[t] = ri[t+o];
      }
      __syncthreads();
    }
    if (t == 0) { idx[(long)b*KEEP_ + it] = ri[0]; vals[ri[0]] = -3.4e38f; }
    __syncthreads();
  }
}

__global__ void pos_out_kernel(const float* __restrict__ pos, const float* __restrict__ pos_part,
                               const int* __restrict__ idx, float* __restrict__ out_pos) {
  const int b = blockIdx.x, t = threadIdx.x;
  if (t >= N3_) return;
  const int src = (t == 0) ? 0 : 1 + idx[(long)b*KEEP_ + t - 1];
  for (int c = 0; c < 2; ++c) {
    float s = 0.f;
    for (int h = 0; h < H_; ++h) s += pos_part[(((long)b*H_ + h)*N_ + src)*2 + c];
    const float pt = s * (1.f / H_);
    out_pos[((long)b*N3_ + t)*2 + c] = (pos[((long)b*N_ + src)*2 + c] + pt) * 0.5f;
  }
}

__global__ void gather_ln2_kernel(const float* __restrict__ x2, const int* __restrict__ idx,
    const float* __restrict__ w, const float* __restrict__ bb,
    float* __restrict__ x3, bf16* __restrict__ x3n) {
  __shared__ float red[256];
  const int row = blockIdx.x;
  const int b = row / N3_, i = row - b * N3_;
  const int src = (i == 0) ? 0 : 1 + idx[(long)b*KEEP_ + i - 1];
  const float* xr = x2 + ((long)b*N_ + src) * C_;
  float s = 0.f;
  for (int c = threadIdx.x; c < C_; c += 256) s += xr[c];
  const float mu = block_sum256(s, red) * (1.f / C_);
  float v = 0.f;
  for (int c = threadIdx.x; c < C_; c += 256) { float d = xr[c] - mu; v += d * d; }
  const float inv = rsqrtf(block_sum256(v, red) * (1.f / C_) + 1e-5f);
  for (int c = threadIdx.x; c < C_; c += 256) {
    const float xv = xr[c];
    x3 [(long)row*C_ + c] = xv;
    x3n[(long)row*C_ + c] = (bf16)((xv - mu) * inv * w[c] + bb[c]);
  }
}

__global__ void __launch_bounds__(128) gemm_fc1_kernel(const bf16* __restrict__ a,
    const bf16* __restrict__ wb, const float* __restrict__ bias, bf16* __restrict__ h1) {
  const int lane = threadIdx.x & 31;
  const int w = blockIdx.x * 4 + (threadIdx.x >> 5);
  const int MT = M3_ / 16;
  if (w >= MT * (HID_ / 64)) return;
  const int mt = w % MT, nt = w / MT;
  const v8f zero = {0,0,0,0,0,0,0,0};
  v8f acc[4] = {zero, zero, zero, zero};
  gemm_core_16x64(a + (long)mt * 16 * C_, C_, wb + (long)nt * 64 * C_, C_, C_, lane, acc);
  const int s = lane >> 4, n15 = lane & 15;
#pragma unroll
  for (int r = 0; r < 8; ++r) {
    const long m = mt * 16 + r + 8 * s;
#pragma unroll
    for (int t = 0; t < 4; ++t) {
      const int n = nt * 64 + t * 16 + n15;
      const float u = acc[t][r] + bias[n];
      h1[m * HID_ + n] = (bf16)(0.5f * u * (1.f + erff(u * 0.70710678118654752f)));
    }
  }
}

__global__ void __launch_bounds__(128) gemm_fc2_kernel(const bf16* __restrict__ h1,
    const bf16* __restrict__ wb, const float* __restrict__ bias,
    const float* __restrict__ x3, float* __restrict__ out) {
  const int lane = threadIdx.x & 31;
  const int w = blockIdx.x * 4 + (threadIdx.x >> 5);
  const int MT = M3_ / 16;
  if (w >= MT * (C_ / 64)) return;
  const int mt = w % MT, nt = w / MT;
  const v8f zero = {0,0,0,0,0,0,0,0};
  v8f acc[4] = {zero, zero, zero, zero};
  gemm_core_16x64(h1 + (long)mt * 16 * HID_, HID_, wb + (long)nt * 64 * HID_, HID_, HID_, lane, acc);
  const int s = lane >> 4, n15 = lane & 15;
#pragma unroll
  for (int r = 0; r < 8; ++r) {
    const long m = mt * 16 + r + 8 * s;
#pragma unroll
    for (int t = 0; t < 4; ++t) {
      const int n = nt * 64 + t * 16 + n15;
      out[m * C_ + n] = acc[t][r] + bias[n] + x3[m * C_ + n];
    }
  }
}

// ---------------------------------------------------------------------------
extern "C" void kernel_launch(void* const* d_in, const int* in_sizes, int n_in,
                              void* d_out, int out_size, void* d_ws, size_t ws_size,
                              hipStream_t stream) {
  (void)in_sizes; (void)n_in; (void)out_size; (void)ws_size;
  const float* x      = (const float*)d_in[0];
  const float* pos    = (const float*)d_in[1];
  const float* ln1_w  = (const float*)d_in[2];
  const float* ln1_b  = (const float*)d_in[3];
  const float* qkv_w  = (const float*)d_in[4];
  const float* gamma  = (const float*)d_in[5];
  const float* proj_w = (const float*)d_in[6];
  const float* proj_b = (const float*)d_in[7];
  const float* ln2_w  = (const float*)d_in[8];
  const float* ln2_b  = (const float*)d_in[9];
  const float* fc1_w  = (const float*)d_in[10];
  const float* fc1_b  = (const float*)d_in[11];
  const float* fc2_w  = (const float*)d_in[12];
  const float* fc2_b  = (const float*)d_in[13];

  char* ws = (char*)d_ws;
  size_t cur = 0;
  auto take = [&](size_t bytes) { size_t r = cur; cur += (bytes + 255) & ~(size_t)255; return r; };

  const size_t QKV_ELT  = (size_t)B_ * H_ * NPAD * HD_;     // q / k / vT elements
  const size_t xn_off   = take((size_t)M1_ * C_ * 2);       // transient
  const size_t q_off    = take(QKV_ELT * 2);                // transient
  const size_t k_off    = take(QKV_ELT * 2);                // transient
  const size_t vt_off   = take(QKV_ELT * 2);                // transient
  const size_t ao_off   = take((size_t)M1_ * C_ * 2);       // transient
  const size_t h1_off   = xn_off;                           // aliases dead transients (79.6MB < 146MB)
  const size_t x2_off   = take((size_t)M1_ * C_ * 4);
  const size_t cls_off  = take((size_t)B_ * H_ * (N_-1) * 4);
  const size_t pp_off   = take((size_t)B_ * H_ * N_ * 2 * 4);
  const size_t mm_off   = take(256);
  const size_t idx_off  = take((size_t)B_ * KEEP_ * 4);
  const size_t x3_off   = take((size_t)M3_ * C_ * 4);
  const size_t x3n_off  = take((size_t)M3_ * C_ * 2);
  const size_t wq_off   = take((size_t)3 * C_ * C_ * 2);
  const size_t wp_off   = take((size_t)C_ * C_ * 2);
  const size_t w1_off   = take((size_t)HID_ * C_ * 2);
  const size_t w2_off   = take((size_t)C_ * HID_ * 2);

  bf16*  xn    = (bf16*) (ws + xn_off);
  bf16*  q     = (bf16*) (ws + q_off);
  bf16*  k     = (bf16*) (ws + k_off);
  bf16*  vT    = (bf16*) (ws + vt_off);
  bf16*  ao    = (bf16*) (ws + ao_off);
  bf16*  h1    = (bf16*) (ws + h1_off);
  float* x2    = (float*)(ws + x2_off);
  float* cls   = (float*)(ws + cls_off);
  float* ppart = (float*)(ws + pp_off);
  float* d_mm  = (float*)(ws + mm_off);
  int*   idx   = (int*)  (ws + idx_off);
  float* x3    = (float*)(ws + x3_off);
  bf16*  x3n   = (bf16*) (ws + x3n_off);
  bf16*  wqkv  = (bf16*) (ws + wq_off);
  bf16*  wproj = (bf16*) (ws + wp_off);
  bf16*  wfc1  = (bf16*) (ws + w1_off);
  bf16*  wfc2  = (bf16*) (ws + w2_off);

  // zero q/k/vT so the NPAD padding contributes 0 through WMMA
  hipMemsetAsync(ws + q_off,  0, QKV_ELT * 2, stream);
  hipMemsetAsync(ws + k_off,  0, QKV_ELT * 2, stream);
  hipMemsetAsync(ws + vt_off, 0, QKV_ELT * 2, stream);

  cvt_bf16_kernel<<<2048, 256, 0, stream>>>(qkv_w,  wqkv,  3 * C_ * C_);
  cvt_bf16_kernel<<<2048, 256, 0, stream>>>(proj_w, wproj, C_ * C_);
  cvt_bf16_kernel<<<2048, 256, 0, stream>>>(fc1_w,  wfc1,  HID_ * C_);
  cvt_bf16_kernel<<<2048, 256, 0, stream>>>(fc2_w,  wfc2,  C_ * HID_);

  ln1_kernel<<<M1_, 256, 0, stream>>>(x, ln1_w, ln1_b, xn);
  gemm_qkv_kernel<<<(M1_/16) * (3*C_/64) / 4, 128, 0, stream>>>(xn, wqkv, gamma, q, k, vT);
  mnmx_kernel<<<B_, 256, 0, stream>>>(pos, d_mm);
  attn_kernel<<<dim3(RT_, H_, B_), 32, 0, stream>>>(q, k, vT, pos, d_mm, gamma, ao, ppart, cls);
  gemm_proj_kernel<<<(M1_/16) * (C_/64) / 4, 128, 0, stream>>>(ao, wproj, proj_b, x, x2);
  topk_kernel<<<B_, 256, 0, stream>>>(cls, idx);
  pos_out_kernel<<<B_, 512, 0, stream>>>(pos, ppart, idx, (float*)d_out + (size_t)M3_ * C_);
  gather_ln2_kernel<<<M3_, 256, 0, stream>>>(x2, idx, ln2_w, ln2_b, x3, x3n);
  gemm_fc1_kernel<<<(M3_/16) * (HID_/64) / 4, 128, 0, stream>>>(x3n, wfc1, fc1_b, h1);
  gemm_fc2_kernel<<<(M3_/16) * (C_/64) / 4, 128, 0, stream>>>(h1, wfc2, fc2_b, x3, (float*)d_out);
}
